// STHD_SpGAT_75814762709187
// MI455X (gfx1250) — matrix operand order repaired and checked by reading, not compiled, not run on hardware
//
#include <hip/hip_runtime.h>
#include <hip/hip_bf16.h>

#define NN 50000
#define KK 32
#define GG 512
#define EE 400000
#define OO 8
#define NTILES (NN/16)   // 3125, exact

typedef __attribute__((ext_vector_type(16))) _Float16 v16h;
typedef __attribute__((ext_vector_type(8)))  float    v8f;

// ---- workspace layout (float offsets) ----
#define OFF_ACC   0                      // 2 floats: [0]=ll_sum, [1]=ce_sum
#define OFF_Q3    4                      // 32 floats
#define OFF_XL    36                     // N*8 floats
#define OFF_XR    (OFF_XL + NN*8)
#define OFF_EV    (OFF_XR + NN*8)        // E floats (e, then ex in-place)
#define OFF_MENC  (OFF_EV + EE)          // N uint32 (encoded segment max)
#define OFF_DEN   (OFF_MENC + NN)        // N floats
#define OFF_BPACK 1300040                // padded to 32B: 5*16*32 v16h fragments

__device__ __forceinline__ float wave_sum(float v) {
  #pragma unroll
  for (int m = 16; m > 0; m >>= 1) v += __shfl_xor(v, m, 32);
  return v;
}
__device__ __forceinline__ float wave_max(float v) {
  #pragma unroll
  for (int m = 16; m > 0; m >>= 1) v = fmaxf(v, __shfl_xor(v, m, 32));
  return v;
}
// monotone float<->uint map for atomicMax on signed floats
__device__ __forceinline__ unsigned fenc(float f) {
  unsigned u = __float_as_uint(f);
  return (u & 0x80000000u) ? ~u : (u | 0x80000000u);
}
__device__ __forceinline__ float fdec(unsigned u) {
  return (u & 0x80000000u) ? __uint_as_float(u & 0x7FFFFFFFu)
                           : __uint_as_float(~u);
}

// ---- init scratch (must run every call; harness does not re-zero) ----
__global__ void k_init(float* ws) {
  int i = blockIdx.x * blockDim.x + threadIdx.x;
  if (i < NN) {
    ((unsigned*)(ws + OFF_MENC))[i] = 0u;   // below fenc(-inf) => identity
    ws[OFF_DEN + i] = 0.f;
  }
  if (i < 2) ws[OFF_ACC + i] = 0.f;
}

// ---- P = softmax(W) row-wise, one wave per row ----
__global__ __launch_bounds__(256) void k_softmax(const float* __restrict__ W,
                                                 float* __restrict__ P) {
  int row  = (blockIdx.x * blockDim.x + threadIdx.x) >> 5;
  int lane = threadIdx.x & 31;
  if (row >= NN) return;
  float x  = W[row * KK + lane];
  float mx = wave_max(x);
  float e  = expf(x - mx);
  float sm = wave_sum(e);
  P[row * KK + lane] = e / sm;
}

// ---- pre-swizzle B fragments into WMMA register layout + compute q3 ----
// B 32x16 f16 k order used here, same as A's (wave32): lane holds column c=lane&15,
// halves t=0..15 hold contraction rows g' = (lane>>4)*16 + t.
__global__ void k_bpack(const float* __restrict__ Mu, const float* __restrict__ Var,
                        const float* __restrict__ Wl, const float* __restrict__ Wr,
                        float* ws) {
  int tid = blockIdx.x * blockDim.x + threadIdx.x;
  _Float16* bp = (_Float16*)(ws + OFF_BPACK);
  if (tid < 512) {                       // 16 steps x 32 lanes
    int step = tid >> 5, lane = tid & 31;
    int c  = lane & 15;
    int gb = step * 32 + (lane >> 4) * 16;
    for (int t = 0; t < 16; ++t) {
      int g = gb + t;
      float iv0 = 1.f / Var[c * GG + g];
      float iv1 = 1.f / Var[(c + 16) * GG + g];
      bp[((0*16 + step)*32 + lane)*16 + t] = (_Float16)iv0;                       // q1 cols 0-15
      bp[((1*16 + step)*32 + lane)*16 + t] = (_Float16)iv1;                       // q1 cols 16-31
      bp[((2*16 + step)*32 + lane)*16 + t] = (_Float16)(Mu[c*GG + g] * iv0);      // q2 cols 0-15
      bp[((3*16 + step)*32 + lane)*16 + t] = (_Float16)(Mu[(c+16)*GG + g] * iv1); // q2 cols 16-31
      bp[((4*16 + step)*32 + lane)*16 + t] =                                      // [Wl | Wr]
          (_Float16)(c < OO ? Wl[g*OO + c] : Wr[g*OO + (c - OO)]);
    }
  }
  if (tid < KK) {                        // q3[k] = sum_g Mu^2/Var
    float s = 0.f;
    for (int g = 0; g < GG; ++g) {
      float mu = Mu[tid * GG + g];
      s += mu * mu / Var[tid * GG + g];
    }
    ws[OFF_Q3 + tid] = s;
  }
}

// ---- fused single pass over X: q1,q2 (WMMA) -> ll partial; xl,xr (WMMA) ----
__global__ __launch_bounds__(128) void k_fused_wmma(
    const float* __restrict__ X, const float* __restrict__ S,
    const float* __restrict__ bl, const float* __restrict__ br,
    const float* __restrict__ P, float* __restrict__ ws) {
  const v16h* frag = (const v16h*)(ws + OFF_BPACK);
  float* XL = ws + OFF_XL;
  float* XR = ws + OFF_XR;
  const float* q3 = ws + OFF_Q3;

  int wave = threadIdx.x >> 5;
  int lane = threadIdx.x & 31;
  int tile = blockIdx.x * 4 + wave;      // wave-uniform guard -> EXEC all-1 at WMMA
  float part = 0.f;

  if (tile < NTILES) {
    int row0 = tile * 16;
    int m  = lane & 15;                  // A-matrix row M
    int hi = lane >> 4;                  // lane half selector
    v8f aq1[2] = {}; v8f aq2[2] = {}; v8f alr = {};
    const float* xrow = X + (size_t)(row0 + m) * GG;

    for (int step = 0; step < 16; ++step) {
      int g0 = step * 32;
      // A fragment (16x32 f16): halves 0-7 -> K = hi*16 + j ; halves 8-15 -> K = hi*16 + 8 + j
      const float* plo = xrow + g0 + hi * 16;
      const float* phi = plo + 8;
      float4 l0 = *(const float4*)plo;
      float4 l1 = *(const float4*)(plo + 4);
      float4 h0 = *(const float4*)phi;
      float4 h1 = *(const float4*)(phi + 4);
      if (step < 15) __builtin_prefetch(xrow + g0 + 32 + hi * 8, 0, 1);
      float lof[8] = {l0.x, l0.y, l0.z, l0.w, l1.x, l1.y, l1.z, l1.w};
      float hif[8] = {h0.x, h0.y, h0.z, h0.w, h1.x, h1.y, h1.z, h1.w};
      v16h ax, ax2;
      #pragma unroll
      for (int j = 0; j < 8; ++j) {
        ax [j]     = (_Float16)lof[j];
        ax2[j]     = (_Float16)(lof[j] * lof[j]);
        ax [8 + j] = (_Float16)hif[j];
        ax2[8 + j] = (_Float16)(hif[j] * hif[j]);
      }
      // pre-swizzled B fragments: one coalesced 32B load each
      v16h b0 = frag[(0*16 + step)*32 + lane];
      v16h b1 = frag[(1*16 + step)*32 + lane];
      v16h b2 = frag[(2*16 + step)*32 + lane];
      v16h b3 = frag[(3*16 + step)*32 + lane];
      v16h b4 = frag[(4*16 + step)*32 + lane];

      aq1[0] = __builtin_amdgcn_wmma_f32_16x16x32_f16(false, ax2, false, b0, (short)0, aq1[0], false, false);
      aq1[1] = __builtin_amdgcn_wmma_f32_16x16x32_f16(false, ax2, false, b1, (short)0, aq1[1], false, false);
      aq2[0] = __builtin_amdgcn_wmma_f32_16x16x32_f16(false, ax,  false, b2, (short)0, aq2[0], false, false);
      aq2[1] = __builtin_amdgcn_wmma_f32_16x16x32_f16(false, ax,  false, b3, (short)0, aq2[1], false, false);
      alr    = __builtin_amdgcn_wmma_f32_16x16x32_f16(false, ax,  false, b4, (short)0, alr,    false, false);
    }

    // C/D layout: lane, vgpr v -> (M = 8*hi + v, N = lane&15)
    int col = lane & 15;
    #pragma unroll
    for (int v = 0; v < 8; ++v) {        // xl/xr + bias -> scratch for edge pass
      int n = row0 + hi * 8 + v;
      float val = alr[v] + (col < OO ? bl[col] : br[col - OO]);
      if (col < OO) XL[n * OO + col]        = val;
      else          XR[n * OO + (col - OO)] = val;
    }
    #pragma unroll
    for (int ct = 0; ct < 2; ++ct) {     // ll_prot partial: sum P * Fmat
      int k = ct * 16 + col;
      float q3v = q3[k];
      #pragma unroll
      for (int v = 0; v < 8; ++v) {
        int n = row0 + hi * 8 + v;
        float sv = S[n];
        float F = -0.5f * (aq1[ct][v] - 2.f * sv * aq2[ct][v] + sv * sv * q3v);
        part += P[n * KK + k] * F;
      }
    }
  }
  part = wave_sum(part);
  __shared__ float sred[4];
  if (lane == 0) sred[wave] = part;
  __syncthreads();
  if (threadIdx.x == 0)
    atomicAdd(ws + OFF_ACC + 0, sred[0] + sred[1] + sred[2] + sred[3]);
}

// ---- edge pass 1: e = leaky_relu(xl[src]+xr[dst]) . att ; segment max ----
__global__ __launch_bounds__(256) void k_edge1(const int* __restrict__ src,
                                               const int* __restrict__ dst,
                                               const float* __restrict__ att,
                                               float* ws) {
  int i = blockIdx.x * blockDim.x + threadIdx.x;
  if (i >= EE) return;
  const float* XL = ws + OFF_XL;
  const float* XR = ws + OFF_XR;
  int s = src[i], d = dst[i];
  float e = 0.f;
  #pragma unroll
  for (int j = 0; j < OO; ++j) {
    float v = XL[s * OO + j] + XR[d * OO + j];
    v = v >= 0.f ? v : 0.2f * v;
    e += v * att[j];
  }
  ws[OFF_EV + i] = e;
  atomicMax((unsigned*)(ws + OFF_MENC) + d, fenc(e));
}

// ---- edge pass 2: ex = exp(e - m[dst]) ; segment sum ----
__global__ __launch_bounds__(256) void k_edge2(const int* __restrict__ dst, float* ws) {
  int i = blockIdx.x * blockDim.x + threadIdx.x;
  if (i >= EE) return;
  int d = dst[i];
  float m  = fdec(((const unsigned*)(ws + OFF_MENC))[d]);
  float ex = expf(ws[OFF_EV + i] - m);
  ws[OFF_EV + i] = ex;
  atomicAdd(ws + OFF_DEN + d, ex);
}

// ---- edge pass 3: ce partial = alpha * (P[src] . log(P[dst]+eps)), wave/edge ----
__global__ __launch_bounds__(256) void k_edge3(const int* __restrict__ src,
                                               const int* __restrict__ dst,
                                               const float* __restrict__ P,
                                               float* ws) {
  int gw   = (blockIdx.x * blockDim.x + threadIdx.x) >> 5;
  int lane = threadIdx.x & 31;
  float t = 0.f;
  int d = 0;
  if (gw < EE) {
    int s = src[gw]; d = dst[gw];
    float ps = P[s * KK + lane];
    float ld = logf(P[d * KK + lane] + 1e-8f);
    t = ps * ld;
  }
  t = wave_sum(t);
  float contrib = 0.f;
  if (gw < EE && lane == 0)
    contrib = (ws[OFF_EV + gw] / ws[OFF_DEN + d]) * t;
  __shared__ float sred[8];
  int w = threadIdx.x >> 5;
  if (lane == 0) sred[w] = contrib;
  __syncthreads();
  if (threadIdx.x == 0) {
    float ssum = 0.f;
    #pragma unroll
    for (int j = 0; j < 8; ++j) ssum += sred[j];
    atomicAdd(ws + OFF_ACC + 1, ssum);
  }
}

// ---- finalize scalars ----
__global__ void k_final(const float* ws, float* out) {
  if (blockIdx.x == 0 && threadIdx.x == 0) {
    out[0] =  ws[OFF_ACC + 0] / (float)NN;   // ll_prot
    out[1] = -ws[OFF_ACC + 1] / (float)NN;   // ce_space
  }
}

extern "C" void kernel_launch(void* const* d_in, const int* in_sizes, int n_in,
                              void* d_out, int out_size, void* d_ws, size_t ws_size,
                              hipStream_t stream) {
  const float* X   = (const float*)d_in[0];
  const float* Mu  = (const float*)d_in[1];
  const float* Var = (const float*)d_in[2];
  const float* W   = (const float*)d_in[3];
  const float* S   = (const float*)d_in[4];
  const float* Wl  = (const float*)d_in[5];
  const float* bl  = (const float*)d_in[6];
  const float* Wr  = (const float*)d_in[7];
  const float* br  = (const float*)d_in[8];
  const float* att = (const float*)d_in[9];
  const int*   ei  = (const int*)d_in[10];
  const int* src = ei;
  const int* dst = ei + EE;

  float* out = (float*)d_out;
  float* P   = out + 2;          // output order: ll_prot, ce_space, P[N*K]
  float* ws  = (float*)d_ws;

  k_init      <<<(NN + 255) / 256, 256, 0, stream>>>(ws);
  k_softmax   <<<(NN * 32 + 255) / 256, 256, 0, stream>>>(W, P);
  k_bpack     <<<2, 256, 0, stream>>>(Mu, Var, Wl, Wr, ws);
  k_fused_wmma<<<(NTILES + 3) / 4, 128, 0, stream>>>(X, S, bl, br, P, ws);
  k_edge1     <<<(EE + 255) / 256, 256, 0, stream>>>(src, dst, att, ws);
  k_edge2     <<<(EE + 255) / 256, 256, 0, stream>>>(dst, ws);
  k_edge3     <<<(EE * 32 + 255) / 256, 256, 0, stream>>>(src, dst, P, ws);
  k_final     <<<1, 1, 0, stream>>>(ws, out);
}
